// MHA_14164802142240
// MI455X (gfx1250) — matrix operand order repaired and checked by reading, not compiled, not run on hardware
//
#include <hip/hip_runtime.h>

typedef __attribute__((ext_vector_type(16))) __bf16 v16bf;
typedef __attribute__((ext_vector_type(8)))  float  v8f;
typedef __attribute__((ext_vector_type(4)))  int    i32x4;

union FragBF {
    uint4 u[2];   // 32 bytes = 16 bf16
    v16bf v;
};

#if defined(__HIP_DEVICE_COMPILE__) && __has_builtin(__builtin_amdgcn_global_load_async_to_lds_b128)
#define HAVE_ASYNC_LDS 1
#else
#define HAVE_ASYNC_LDS 0
#endif

#if HAVE_ASYNC_LDS
typedef __attribute__((address_space(1))) i32x4 glb_i32x4;   // global (device) int4
typedef __attribute__((address_space(3))) i32x4 lds_i32x4;   // LDS (shared) int4

__device__ __forceinline__ glb_i32x4* as_glb(const void* p) {
    return (glb_i32x4*)(unsigned long long)p;
}
__device__ __forceinline__ lds_i32x4* as_lds(void* p) {
    // generic LDS pointer carries the LDS offset in its low 32 bits
    return (lds_i32x4*)(unsigned int)(unsigned long long)p;
}
__device__ __forceinline__ void wait_async0() {
#if __has_builtin(__builtin_amdgcn_s_wait_asynccnt)
    __builtin_amdgcn_s_wait_asynccnt(0);
#else
    asm volatile("s_wait_asynccnt 0x0" ::: "memory");
#endif
}
#endif

__device__ __forceinline__ unsigned short f2bf(float f) {
    unsigned int u = __float_as_uint(f);
    unsigned int r = u + 0x7FFFu + ((u >> 16) & 1u);   // round-to-nearest-even
    return (unsigned short)(r >> 16);
}

// ---------------------------------------------------------------------------
// f32 -> bf16 convert
// ---------------------------------------------------------------------------
__global__ void f32_to_bf16_kernel(const float* __restrict__ in,
                                   unsigned short* __restrict__ out, int n) {
    int i = blockIdx.x * blockDim.x + threadIdx.x;
    if (i < n) out[i] = f2bf(in[i]);
}

// ---------------------------------------------------------------------------
// C[M,N] (f32) = A[M,K] (bf16) * B[N,K]^T (bf16)   "NT" GEMM
// Block = 256 threads (8 waves), tile 128x128, BK = 32, double-buffered LDS.
// Data movement: GLOBAL_LOAD_ASYNC_TO_LDS_B128 (ASYNCcnt) when available.
// Wave w: rows (w/2)*32 .. +31 (2 mtiles), cols (w%2)*64 .. +63 (4 ntiles).
// ---------------------------------------------------------------------------
__global__ __launch_bounds__(256)
void gemm_nt_bf16(const unsigned short* __restrict__ A,
                  const unsigned short* __restrict__ B,
                  float* __restrict__ C,
                  int M, int N, int K) {
    __shared__ unsigned short As[2][128][40];   // stride 40 elems = 80B (16B mult)
    __shared__ unsigned short Bs[2][128][40];

    const int tid  = threadIdx.x;
    const int wave = tid >> 5;
    const int lane = tid & 31;
    const int lo   = lane & 15;
    const int hi   = lane >> 4;

    const int m0 = blockIdx.y * 128;
    const int n0 = blockIdx.x * 128;
    const int wm = (wave >> 1) * 32;
    const int wn = (wave & 1) * 64;

    v8f acc[2][4];
#pragma unroll
    for (int mt = 0; mt < 2; ++mt)
#pragma unroll
        for (int nt = 0; nt < 4; ++nt) acc[mt][nt] = v8f{};

    const int NTK = K >> 5;   // number of 32-wide K tiles

    auto compute_tile = [&](int buf) {
        FragBF af[2], bfrag[4];
#pragma unroll
        for (int mt = 0; mt < 2; ++mt) {
            int r = wm + mt * 16 + lo;
            af[mt].u[0] = *(const uint4*)&As[buf][r][hi * 8];       // K: hi*8..+7
            af[mt].u[1] = *(const uint4*)&As[buf][r][16 + hi * 8];  // K: 16+hi*8..+7
        }
#pragma unroll
        for (int nt = 0; nt < 4; ++nt) {
            int r = wn + nt * 16 + lo;
            bfrag[nt].u[0] = *(const uint4*)&Bs[buf][r][hi * 8];      // K: hi*8..+7
            bfrag[nt].u[1] = *(const uint4*)&Bs[buf][r][16 + hi * 8]; // K: 16+hi*8..+7
        }
#pragma unroll
        for (int mt = 0; mt < 2; ++mt)
#pragma unroll
            for (int nt = 0; nt < 4; ++nt)
                acc[mt][nt] = __builtin_amdgcn_wmma_f32_16x16x32_bf16(
                    false, af[mt].v, false, bfrag[nt].v, (short)0, acc[mt][nt],
                    false, false);
    };

#if HAVE_ASYNC_LDS
    auto issue_tile = [&](int kt, int buf) {
#pragma unroll
        for (int i = 0; i < 2; ++i) {
            int g  = tid + i * 256;      // 0..511 groups of 8 elems
            int r  = g >> 2;             // 0..127
            int cg = g & 3;              // 0..3
            __builtin_amdgcn_global_load_async_to_lds_b128(
                as_glb(A + (size_t)(m0 + r) * K + kt * 32 + cg * 8),
                as_lds(&As[buf][r][cg * 8]), 0, 0);
            __builtin_amdgcn_global_load_async_to_lds_b128(
                as_glb(B + (size_t)(n0 + r) * K + kt * 32 + cg * 8),
                as_lds(&Bs[buf][r][cg * 8]), 0, 0);
        }
    };

    issue_tile(0, 0);
    wait_async0();
    __syncthreads();
    for (int kt = 0; kt < NTK; ++kt) {
        const int cur = kt & 1;
        if (kt + 1 < NTK) issue_tile(kt + 1, cur ^ 1);  // overlap next transfer
        compute_tile(cur);
        wait_async0();        // this wave's async writes into (cur^1) landed
        __syncthreads();      // everyone's landed; readers of cur are done
    }
#else
    // Fallback: synchronous staging through VGPRs (single buffer).
    for (int kt = 0; kt < NTK; ++kt) {
        uint4 ar[2], br[2];
#pragma unroll
        for (int i = 0; i < 2; ++i) {
            int g  = tid + i * 256;
            int r  = g >> 2;
            int cg = g & 3;
            ar[i] = *(const uint4*)(A + (size_t)(m0 + r) * K + kt * 32 + cg * 8);
            br[i] = *(const uint4*)(B + (size_t)(n0 + r) * K + kt * 32 + cg * 8);
        }
        __syncthreads();
#pragma unroll
        for (int i = 0; i < 2; ++i) {
            int g  = tid + i * 256;
            int r  = g >> 2;
            int cg = g & 3;
            *(uint4*)&As[0][r][cg * 8] = ar[i];
            *(uint4*)&Bs[0][r][cg * 8] = br[i];
        }
        __syncthreads();
        compute_tile(0);
    }
#endif

#pragma unroll
    for (int mt = 0; mt < 2; ++mt)
#pragma unroll
        for (int nt = 0; nt < 4; ++nt) {
            int col   = n0 + wn + nt * 16 + lo;
            int rbase = m0 + wm + mt * 16 + 8 * hi;
#pragma unroll
            for (int j = 0; j < 8; ++j)
                C[(size_t)(rbase + j) * N + col] = acc[mt][nt][j];
        }
}

// ---------------------------------------------------------------------------
// RoPE + split qkv f32 [B,S,3E] -> q,k,v bf16 [B,H,S,D], q pre-scaled 1/sqrt(D)
// B=2, S=2048, E=1024, H=16, D=64. One thread per (b,s,h,pair i<32).
// ---------------------------------------------------------------------------
__global__ void rope_split_kernel(const float* __restrict__ qkv,
                                  unsigned short* __restrict__ qb,
                                  unsigned short* __restrict__ kb,
                                  unsigned short* __restrict__ vb) {
    int idx = blockIdx.x * blockDim.x + threadIdx.x;  // 2^21 threads
    int i = idx & 31;
    int h = (idx >> 5) & 15;
    int s = (idx >> 9) & 2047;
    int b = idx >> 20;

    const size_t row = (size_t)(b * 2048 + s) * 3072;
    const int    d2  = h * 64 + 2 * i;
    float q0 = qkv[row + d2],        q1 = qkv[row + d2 + 1];
    float k0 = qkv[row + 1024 + d2], k1 = qkv[row + 1024 + d2 + 1];
    float v0 = qkv[row + 2048 + d2], v1 = qkv[row + 2048 + d2 + 1];

    // theta = 10000^(-2i/64) = exp(-i * ln(10000)/32)
    float theta = __expf(-(float)i * (9.210340371976184f / 32.0f));
    float ang   = (float)s * theta;
    float c  = cosf(ang);
    float sn = sinf(ang);

    float qo0 = q0 * c - q1 * sn, qo1 = q0 * sn + q1 * c;
    float ko0 = k0 * c - k1 * sn, ko1 = k0 * sn + k1 * c;

    size_t o = ((size_t)(b * 16 + h) * 2048 + s) * 64 + 2 * i;
    qb[o]     = f2bf(qo0 * 0.125f);   // fold 1/sqrt(64)
    qb[o + 1] = f2bf(qo1 * 0.125f);
    kb[o]     = f2bf(ko0);
    kb[o + 1] = f2bf(ko1);
    vb[o]     = f2bf(v0);
    vb[o + 1] = f2bf(v1);
}

// ---------------------------------------------------------------------------
// Causal flash attention. One block = 128 threads (4 waves) per (qtile, h, b).
// Each wave owns a 16-row query strip; keys consumed in 64-wide chunks.
// Q/K fragments straight from global (D contiguous). V transposed into LDS
// per chunk for PV B-fragments; P bf16 round-trips LDS to reach A-layout.
// Next chunk's K/V prefetched (global_prefetch_b8).
// Output: bf16 y in [B,S,E] for the output projection GEMM.
// ---------------------------------------------------------------------------
__global__ __launch_bounds__(128)
void attn_kernel(const unsigned short* __restrict__ Q,
                 const unsigned short* __restrict__ Kt,
                 const unsigned short* __restrict__ V,
                 unsigned short* __restrict__ Y) {
    const int S = 2048, H = 16;
    const int qtile = blockIdx.x;
    const int h     = blockIdx.y;
    const int b     = blockIdx.z;
    const int tid   = threadIdx.x;
    const int wave  = tid >> 5;
    const int lane  = tid & 31;
    const int lo    = lane & 15;
    const int hi    = lane >> 4;

    __shared__ unsigned short Vt[64][72];       // [d][key], stride 144B (16B mult)
    __shared__ unsigned short Pl[4][16][72];    // per-wave P tile

    const size_t bh = (size_t)(b * H + h) * S * 64;
    const unsigned short* q = Q + bh;
    const unsigned short* k = Kt + bh;
    const unsigned short* v = V + bh;

    const int q0    = qtile * 64;
    const int strip = q0 + wave * 16;

    // Hoisted Q A-fragments (rows strip+lo, K-dim = head dim 64 -> 2 frags)
    FragBF qa[2];
    {
        const unsigned short* qr = q + (size_t)(strip + lo) * 64;
#pragma unroll
        for (int kk = 0; kk < 2; ++kk) {
            int d0 = kk * 32;
            qa[kk].u[0] = *(const uint4*)(qr + d0 + hi * 8);
            qa[kk].u[1] = *(const uint4*)(qr + d0 + 16 + hi * 8);
        }
    }

    v8f O[4];
#pragma unroll
    for (int nt = 0; nt < 4; ++nt) O[nt] = v8f{};
    float mrow[8], lrow[8];
#pragma unroll
    for (int j = 0; j < 8; ++j) { mrow[j] = -__builtin_inff(); lrow[j] = 0.0f; }

    const int nch = qtile + 1;
    for (int ch = 0; ch < nch; ++ch) {
        const int kb0 = ch * 64;

        // prefetch next chunk's K/V rows while working on this one
        if (ch + 1 < nch) {
            const size_t nrow = (size_t)(kb0 + 64 + (tid & 63)) * 64;
            __builtin_prefetch(k + nrow, 0, 1);
            __builtin_prefetch(v + nrow, 0, 1);
        }

        __syncthreads();   // protect Vt reuse
        // cooperative transpose of 64x64 V chunk into Vt[d][key]
#pragma unroll
        for (int it = 0; it < 4; ++it) {
            int g   = it * 128 + tid;   // 0..511
            int key = g & 63;
            int db  = g >> 6;           // 0..7
            uint4 pack = *(const uint4*)(v + (size_t)(kb0 + key) * 64 + db * 8);
            const unsigned short* pw = (const unsigned short*)&pack;
#pragma unroll
            for (int e = 0; e < 8; ++e) Vt[db * 8 + e][key] = pw[e];
        }
        __syncthreads();

        // S = Q * K^T (K rows straight from global as B-fragments)
        v8f st[4];
#pragma unroll
        for (int nt = 0; nt < 4; ++nt) {
            v8f c = v8f{};
            const unsigned short* kr = k + (size_t)(kb0 + nt * 16 + lo) * 64;
#pragma unroll
            for (int kk = 0; kk < 2; ++kk) {
                FragBF kf;
                int d0 = kk * 32;
                kf.u[0] = *(const uint4*)(kr + d0 + hi * 8);
                kf.u[1] = *(const uint4*)(kr + d0 + 16 + hi * 8);
                c = __builtin_amdgcn_wmma_f32_16x16x32_bf16(
                    false, qa[kk].v, false, kf.v, (short)0, c, false, false);
            }
            st[nt] = c;
        }

        // causal mask + chunk row max
        float cmax[8];
#pragma unroll
        for (int j = 0; j < 8; ++j) {
            int   qrow = strip + j + 8 * hi;
            float mx   = -__builtin_inff();
#pragma unroll
            for (int nt = 0; nt < 4; ++nt) {
                int   kcol = kb0 + nt * 16 + lo;
                float sv   = st[nt][j];
                if (kcol > qrow) { sv = -__builtin_inff(); st[nt][j] = sv; }
                mx = fmaxf(mx, sv);
            }
            cmax[j] = mx;
        }
#pragma unroll
        for (int m = 1; m < 16; m <<= 1)
#pragma unroll
            for (int j = 0; j < 8; ++j)
                cmax[j] = fmaxf(cmax[j], __shfl_xor(cmax[j], m, 32));

        // online softmax update + write P (bf16) to per-wave LDS tile
        float psum[8];
#pragma unroll
        for (int j = 0; j < 8; ++j) {
            float mnew  = fmaxf(mrow[j], cmax[j]);
            float scale = __expf(mrow[j] - mnew);
            mrow[j] = mnew;
#pragma unroll
            for (int nt = 0; nt < 4; ++nt) O[nt][j] *= scale;
            lrow[j] *= scale;
            float ps = 0.0f;
#pragma unroll
            for (int nt = 0; nt < 4; ++nt) {
                float p = __expf(st[nt][j] - mnew);
                ps += p;
                Pl[wave][j + 8 * hi][nt * 16 + lo] = f2bf(p);
            }
            psum[j] = ps;
        }
#pragma unroll
        for (int m = 1; m < 16; m <<= 1)
#pragma unroll
            for (int j = 0; j < 8; ++j)
                psum[j] += __shfl_xor(psum[j], m, 32);
#pragma unroll
        for (int j = 0; j < 8; ++j) lrow[j] += psum[j];

        // O += P * V  (P as A-frags from LDS, V^T as B-frags from LDS)
        FragBF pa[2];
#pragma unroll
        for (int kk = 0; kk < 2; ++kk) {
            int d0 = kk * 32;
            pa[kk].u[0] = *(const uint4*)&Pl[wave][lo][d0 + hi * 8];
            pa[kk].u[1] = *(const uint4*)&Pl[wave][lo][d0 + 16 + hi * 8];
        }
#pragma unroll
        for (int nt = 0; nt < 4; ++nt) {
#pragma unroll
            for (int kk = 0; kk < 2; ++kk) {
                FragBF vf;
                int d0 = kk * 32;
                vf.u[0] = *(const uint4*)&Vt[nt * 16 + lo][d0 + hi * 8];
                vf.u[1] = *(const uint4*)&Vt[nt * 16 + lo][d0 + 16 + hi * 8];
                O[nt] = __builtin_amdgcn_wmma_f32_16x16x32_bf16(
                    false, pa[kk].v, false, vf.v, (short)0, O[nt], false, false);
            }
        }
    }

    // normalize and store y (bf16, [B,S,E] with E = h*64 + d)
#pragma unroll
    for (int nt = 0; nt < 4; ++nt) {
#pragma unroll
        for (int j = 0; j < 8; ++j) {
            int   qrow = strip + j + 8 * hi;
            float val  = O[nt][j] / lrow[j];
            Y[(size_t)(b * S + qrow) * 1024 + h * 64 + nt * 16 + lo] = f2bf(val);
        }
    }
}

// ---------------------------------------------------------------------------
// Host-side orchestration
// ---------------------------------------------------------------------------
extern "C" void kernel_launch(void* const* d_in, const int* in_sizes, int n_in,
                              void* d_out, int out_size, void* d_ws, size_t ws_size,
                              hipStream_t stream) {
    const float* x    = (const float*)d_in[0];   // [2,2048,1024]
    const float* wqkv = (const float*)d_in[1];   // [3072,1024]
    const float* wo   = (const float*)d_in[2];   // [1024,1024]
    float*       out  = (float*)d_out;           // [2,2048,1024]

    const int Bb = 2, S = 2048, E = 1024, H = 16;
    const int M = Bb * S;                        // 4096

    char*  ws  = (char*)d_ws;
    size_t off = 0;
    auto carve = [&](size_t bytes) -> void* {
        void* p = ws + off;
        off = (off + bytes + 255) & ~(size_t)255;
        return p;
    };

    unsigned short* xb    = (unsigned short*)carve((size_t)M * E * 2);
    unsigned short* wqkvb = (unsigned short*)carve((size_t)3 * E * E * 2);
    unsigned short* wob   = (unsigned short*)carve((size_t)E * E * 2);
    float*          qkvf  = (float*)carve((size_t)M * 3 * E * 4);
    unsigned short* qb    = (unsigned short*)carve((size_t)M * E * 2);
    unsigned short* kb    = (unsigned short*)carve((size_t)M * E * 2);
    unsigned short* vb    = (unsigned short*)carve((size_t)M * E * 2);
    unsigned short* yb    = (unsigned short*)carve((size_t)M * E * 2);

    // 1) convert inputs to bf16
    f32_to_bf16_kernel<<<(M * E) / 256, 256, 0, stream>>>(x, xb, M * E);
    f32_to_bf16_kernel<<<(3 * E * E) / 256, 256, 0, stream>>>(wqkv, wqkvb, 3 * E * E);
    f32_to_bf16_kernel<<<(E * E) / 256, 256, 0, stream>>>(wo, wob, E * E);

    // 2) qkv = x @ W_qkv^T   (M=4096, N=3072, K=1024)
    gemm_nt_bf16<<<dim3(3 * E / 128, M / 128), 256, 0, stream>>>(
        xb, wqkvb, qkvf, M, 3 * E, E);

    // 3) RoPE + split to head-major bf16 q,k,v
    rope_split_kernel<<<(M * 512) / 256, 256, 0, stream>>>(qkvf, qb, kb, vb);

    // 4) causal flash attention -> y bf16 [B,S,E]
    attn_kernel<<<dim3(S / 64, H, Bb), 128, 0, stream>>>(qb, kb, vb, yb);

    // 5) out = y @ W_o^T   (M=4096, N=1024, K=1024)
    gemm_nt_bf16<<<dim3(E / 128, M / 128), 256, 0, stream>>>(
        yb, wob, out, M, E, E);
}
